// Attention_63385127354956
// MI455X (gfx1250) — matrix operand order repaired and checked by reading, not compiled, not run on hardware
//
#include <hip/hip_runtime.h>
#include <hip/hip_bf16.h>
#include <stdint.h>

typedef __attribute__((ext_vector_type(16))) _Float16 v16h;
typedef __attribute__((ext_vector_type(8)))  float    v8f;

#define BH_N     32
#define S_LEN    1024
#define D_DIM    64
#define TK       64      // keys per LDS tile
#define QROWS    64      // Q rows per workgroup (4 waves x 16)
#define KT_STRIDE 68     // halves per feature-row of K^T tile (64 keys + pad, 136B)
#define V_STRIDE  68     // halves per key-row of V tile (64 d + pad)
#define P_STRIDE  72     // halves per row of P staging (64 keys + pad, 144B = 16B aligned)

__global__ __launch_bounds__(128)
void fattn_wmma_kernel(const float* __restrict__ Q, const float* __restrict__ K,
                       const float* __restrict__ V, const uint8_t* __restrict__ mask,
                       float* __restrict__ O)
{
    __shared__ _Float16 ldsKT[D_DIM * KT_STRIDE];   // K tile transposed: [feat][key]
    __shared__ _Float16 ldsV [TK * V_STRIDE];       // V tile row-major:  [key][d]
    __shared__ _Float16 ldsP [4][16 * P_STRIDE];    // per-wave P staging: [m][key]

    const int tid  = threadIdx.x;
    const int lane = tid & 31;
    const int wave = tid >> 5;
    const int lm   = lane & 15;
    const int hi   = lane >> 4;

    const int bh    = blockIdx.x;                   // 0..31
    const int qblk  = blockIdx.y;                   // 0..15
    const int qrow0 = qblk * QROWS + wave * 16;     // this wave's first Q row

    const float*   qb = Q + ((size_t)bh * S_LEN + qrow0) * D_DIM;
    const float*   kb = K + (size_t)bh * S_LEN * D_DIM;
    const float*   vb = V + (size_t)bh * S_LEN * D_DIM;
    const uint8_t* mb = mask + (size_t)bh * S_LEN * S_LEN + (size_t)qrow0 * S_LEN;

    const float neg_inf = -__builtin_inff();
    const float scale   = 0.125f;                   // 1/sqrt(64)

    // ---- Q A-fragments: rows qrow0..+15, features split into two 32-chunks.
    // A layout: lane(0-15)->M, hi-lanes carry upper K halves:
    //   h<8  -> K = h + 8*hi        (+32*c)
    //   h>=8 -> K = (h-8) + 16 + 8*hi (+32*c)
    v16h qa[2];
    {
        const float* qrow = qb + (size_t)lm * D_DIM;
        #pragma unroll
        for (int c = 0; c < 2; ++c) {
            const float4 f0a = ((const float4*)(qrow + c*32 + hi*8))[0];
            const float4 f0b = ((const float4*)(qrow + c*32 + hi*8))[1];
            const float4 f1a = ((const float4*)(qrow + c*32 + 16 + hi*8))[0];
            const float4 f1b = ((const float4*)(qrow + c*32 + 16 + hi*8))[1];
            qa[c][0]=(_Float16)f0a.x;  qa[c][1]=(_Float16)f0a.y;
            qa[c][2]=(_Float16)f0a.z;  qa[c][3]=(_Float16)f0a.w;
            qa[c][4]=(_Float16)f0b.x;  qa[c][5]=(_Float16)f0b.y;
            qa[c][6]=(_Float16)f0b.z;  qa[c][7]=(_Float16)f0b.w;
            qa[c][8]=(_Float16)f1a.x;  qa[c][9]=(_Float16)f1a.y;
            qa[c][10]=(_Float16)f1a.z; qa[c][11]=(_Float16)f1a.w;
            qa[c][12]=(_Float16)f1b.x; qa[c][13]=(_Float16)f1b.y;
            qa[c][14]=(_Float16)f1b.z; qa[c][15]=(_Float16)f1b.w;
        }
    }

    // ---- online-softmax state (per lane: 8 rows, m = j + 8*hi, replicated over 16 lanes)
    float m_i[8], l_i[8];
    v8f   o[4] = {};                                // out accumulators, 4 d-tiles of 16 cols
    #pragma unroll
    for (int j = 0; j < 8; ++j) { m_i[j] = neg_inf; l_i[j] = 0.0f; }

    for (int t = 0; t < S_LEN / TK; ++t) {
        const int key0 = t * TK;
        __syncthreads();                            // protect LDS reuse across iterations

        // ---- cooperative stage: K tile (transposed) and V tile (row-major), fp32->f16
        {
            const int key = tid >> 1;               // 0..63
            const int f0  = (tid & 1) * 32;         // 0 or 32
            const float* ksrc = kb + (size_t)(key0 + key) * D_DIM + f0;
            const float* vsrc = vb + (size_t)(key0 + key) * D_DIM + f0;
            _Float16*    vdst = &ldsV[key * V_STRIDE + f0];
            #pragma unroll
            for (int i = 0; i < 8; ++i) {
                float4 kq = ((const float4*)ksrc)[i];
                int f = f0 + i * 4;
                ldsKT[(f+0)*KT_STRIDE + key] = (_Float16)kq.x;
                ldsKT[(f+1)*KT_STRIDE + key] = (_Float16)kq.y;
                ldsKT[(f+2)*KT_STRIDE + key] = (_Float16)kq.z;
                ldsKT[(f+3)*KT_STRIDE + key] = (_Float16)kq.w;
                float4 vq = ((const float4*)vsrc)[i];
                vdst[i*4+0] = (_Float16)vq.x;
                vdst[i*4+1] = (_Float16)vq.y;
                vdst[i*4+2] = (_Float16)vq.z;
                vdst[i*4+3] = (_Float16)vq.w;
            }
        }
        __syncthreads();

        // ---- scores: S = Q * K^T, 4 n-tiles x 2 k-steps of v_wmma_f32_16x16x32_f16
        // B layout: lane%16 -> N (key), halves -> K(=feature) in table order (+32*c)
        v8f s[4];
        #pragma unroll
        for (int nt = 0; nt < 4; ++nt) {
            v8f acc = {};
            #pragma unroll
            for (int c = 0; c < 2; ++c) {
                v16h bfrag;
                const _Float16* src = &ldsKT[(c*32 + hi*8) * KT_STRIDE + nt*16 + lm];
                #pragma unroll
                for (int h = 0; h < 16; ++h) bfrag[h] = src[(h + (h >= 8 ? 8 : 0)) * KT_STRIDE];
                acc = __builtin_amdgcn_wmma_f32_16x16x32_f16(
                          false, qa[c], false, bfrag, (short)0, acc, false, false);
            }
            s[nt] = acc;
        }

        // ---- scale + mask (C layout: element j of s[nt] is row m=j+8*hi, key nt*16+lm)
        #pragma unroll
        for (int nt = 0; nt < 4; ++nt) {
            const uint8_t* mrow = mb + (size_t)(hi * 8) * S_LEN + key0 + nt*16 + lm;
            #pragma unroll
            for (int j = 0; j < 8; ++j) {
                float sv = s[nt][j] * scale;
                if (mrow[(size_t)j * S_LEN]) sv = neg_inf;
                s[nt][j] = sv;
            }
        }

        // ---- online softmax (row reductions across the 16-lane half-groups)
        #pragma unroll
        for (int j = 0; j < 8; ++j) {
            float tmax = fmaxf(fmaxf(s[0][j], s[1][j]), fmaxf(s[2][j], s[3][j]));
            #pragma unroll
            for (int off = 8; off >= 1; off >>= 1)
                tmax = fmaxf(tmax, __shfl_xor(tmax, off, 32));
            float mnew  = fmaxf(m_i[j], tmax);
            float alpha = (m_i[j] == neg_inf) ? 0.0f : __expf(m_i[j] - mnew);
            float rsum  = 0.0f;
            #pragma unroll
            for (int nt = 0; nt < 4; ++nt) {
                float sv = s[nt][j];
                float p  = (sv == neg_inf) ? 0.0f : __expf(sv - mnew);
                s[nt][j] = p;
                rsum += p;
            }
            #pragma unroll
            for (int off = 8; off >= 1; off >>= 1)
                rsum += __shfl_xor(rsum, off, 32);
            m_i[j] = mnew;
            l_i[j] = l_i[j] * alpha + rsum;
            #pragma unroll
            for (int d = 0; d < 4; ++d) o[d][j] *= alpha;
        }

        // ---- P -> per-wave LDS staging (C layout -> memory [m][key]); wave-private,
        // DS ops from the same wave are in-order so no workgroup barrier needed.
        #pragma unroll
        for (int nt = 0; nt < 4; ++nt) {
            #pragma unroll
            for (int j = 0; j < 8; ++j)
                ldsP[wave][(j + 8*hi) * P_STRIDE + nt*16 + lm] = (_Float16)s[nt][j];
        }

        // ---- O += P * V : 2 k-steps (keys) x 4 d-tiles
        #pragma unroll
        for (int ks = 0; ks < 2; ++ks) {
            v16h pa;                                // A frag of P (re-read in A layout)
            const _Float16* pbase = &ldsP[wave][lm * P_STRIDE + ks*32 + hi*8];
            #pragma unroll
            for (int h = 0; h < 8; ++h) pa[h]     = pbase[h];
            #pragma unroll
            for (int h = 0; h < 8; ++h) pa[8 + h] = pbase[16 + h];
            #pragma unroll
            for (int d = 0; d < 4; ++d) {
                v16h vfrag;                         // B frag: lane%16 -> d-col, halves -> key (table order)
                const _Float16* src = &ldsV[(ks*32 + hi*8) * V_STRIDE + d*16 + lm];
                #pragma unroll
                for (int h = 0; h < 16; ++h) vfrag[h] = src[(h + (h >= 8 ? 8 : 0)) * V_STRIDE];
                o[d] = __builtin_amdgcn_wmma_f32_16x16x32_f16(
                           false, pa, false, vfrag, (short)0, o[d], false, false);
            }
        }
    }

    // ---- epilogue: normalize and store (coalesced: 16 lanes cover 16 consecutive d)
    float* ob = O + ((size_t)bh * S_LEN + qrow0) * D_DIM;
    #pragma unroll
    for (int j = 0; j < 8; ++j) {
        float inv = 1.0f / l_i[j];
        const int m = j + 8 * hi;
        #pragma unroll
        for (int d = 0; d < 4; ++d)
            ob[(size_t)m * D_DIM + d*16 + lm] = o[d][j] * inv;
    }
}

extern "C" void kernel_launch(void* const* d_in, const int* in_sizes, int n_in,
                              void* d_out, int out_size, void* d_ws, size_t ws_size,
                              hipStream_t stream) {
    (void)in_sizes; (void)n_in; (void)out_size; (void)d_ws; (void)ws_size;
    const float*   q  = (const float*)d_in[0];
    const float*   k  = (const float*)d_in[1];
    const float*   v  = (const float*)d_in[2];
    const uint8_t* mk = (const uint8_t*)d_in[3];
    float*         o  = (float*)d_out;
    dim3 grid(BH_N, S_LEN / QROWS);                 // 32 x 16 workgroups
    fattn_wmma_kernel<<<grid, 128, 0, stream>>>(q, k, v, mk, o);
}
